// InnerGPTAttention_49890340110645
// MI455X (gfx1250) — matrix operand order repaired and checked by reading, not compiled, not run on hardware
//
#include <hip/hip_runtime.h>

// ---------------------------------------------------------------------------
// CDNA5 (gfx1250) GPT attention block: QKV GEMM -> flash attention -> out proj
// All matmuls on v_wmma_f32_16x16x32_bf16 (wave32).
// ---------------------------------------------------------------------------

typedef __attribute__((ext_vector_type(8)))  __bf16 v8bf;
typedef __attribute__((ext_vector_type(16))) __bf16 v16bf;
typedef __attribute__((ext_vector_type(8)))  float  v8f;

#define D_MODEL 2048
#define SEQ     2048
#define NHEADS  16
#define DHEAD   128

__device__ __forceinline__ unsigned short f2bf(float f) {
    unsigned int u = __float_as_uint(f);
    u += 0x7FFFu + ((u >> 16) & 1u);          // round-to-nearest-even
    return (unsigned short)(u >> 16);
}

// Assemble a 16-element bf16 fragment from two 16-byte aligned chunks.
__device__ __forceinline__ v16bf ld_frag(const unsigned short* p0,
                                         const unsigned short* p1) {
    v8bf lo = *reinterpret_cast<const v8bf*>(p0);
    v8bf hi = *reinterpret_cast<const v8bf*>(p1);
    v16bf r;
#pragma unroll
    for (int i = 0; i < 8; ++i) { r[i] = lo[i]; r[i + 8] = hi[i]; }
    return r;
}

__device__ __forceinline__ v8f vzero8() {
    v8f v;
#pragma unroll
    for (int i = 0; i < 8; ++i) v[i] = 0.0f;
    return v;
}

#define WMMA_BF16(a, b, c) \
    __builtin_amdgcn_wmma_f32_16x16x32_bf16(false, (a), false, (b), (short)0, (c), false, false)

// ---------------------------------------------------------------------------
// f32 -> bf16 bulk convert (n divisible by 4)
// ---------------------------------------------------------------------------
__global__ __launch_bounds__(256) void cvt_bf16_kernel(const float* __restrict__ in,
                                                       unsigned short* __restrict__ out,
                                                       int n) {
    int i = (blockIdx.x * blockDim.x + threadIdx.x) * 4;
    if (i < n) {
        float4 v = *reinterpret_cast<const float4*>(in + i);
        out[i + 0] = f2bf(v.x);
        out[i + 1] = f2bf(v.y);
        out[i + 2] = f2bf(v.z);
        out[i + 3] = f2bf(v.w);
    }
}

// ---------------------------------------------------------------------------
// out[c*R + r] = bf16(in[r*C + c])   (R, C multiples of 32)
// ---------------------------------------------------------------------------
__global__ __launch_bounds__(256) void transpose_bf16_kernel(const float* __restrict__ in,
                                                             unsigned short* __restrict__ out,
                                                             int R, int C) {
    __shared__ float tile[32][33];
    const int tx = threadIdx.x & 31;
    const int ty = threadIdx.x >> 5;      // 0..7
    const int c0 = blockIdx.x * 32;
    const int r0 = blockIdx.y * 32;
#pragma unroll
    for (int i = 0; i < 32; i += 8)
        tile[ty + i][tx] = in[(size_t)(r0 + ty + i) * C + (c0 + tx)];
    __syncthreads();
#pragma unroll
    for (int i = 0; i < 32; i += 8)
        out[(size_t)(c0 + ty + i) * R + (r0 + tx)] = f2bf(tile[tx][ty + i]);
}

// ---------------------------------------------------------------------------
// QKV GEMM: C[2048,6144] = xb[2048,2048] @ Wqkv + b.  B given as W^T [6144,2048].
// Q,K (cols 0..4095) -> qk[row,4096] bf16 row-major; V (cols 4096..6143) -> vt
// stored TRANSPOSED as [feature, seq] so PV-GEMM B fragments are contiguous.
// Block = 128x128 tile, 8 waves, each wave 2x4 WMMA tiles (32x64).
// The QK/V split at col 4096 is block-uniform (128 | 4096), so the epilogue
// branch is uniform -> no per-element exec-mask juggling.
// ---------------------------------------------------------------------------
__global__ __launch_bounds__(256) void qkv_gemm_kernel(const unsigned short* __restrict__ xb,
                                                       const unsigned short* __restrict__ wt,
                                                       const float* __restrict__ bias,
                                                       unsigned short* __restrict__ qk,
                                                       unsigned short* __restrict__ vt) {
    const int lane  = threadIdx.x & 31;
    const int wave  = threadIdx.x >> 5;
    const int l16   = lane & 15;
    const int hh    = lane >> 4;
    const int waveM = wave >> 1;          // 0..3
    const int waveN = wave & 1;           // 0..1
    const int mbase = blockIdx.y * 128 + waveM * 32;
    const int nbase = blockIdx.x * 128 + waveN * 64;

    const unsigned short* arow[2];
    const unsigned short* brow[4];
#pragma unroll
    for (int mt = 0; mt < 2; ++mt)
        arow[mt] = xb + (size_t)(mbase + mt * 16 + l16) * D_MODEL;
#pragma unroll
    for (int nt = 0; nt < 4; ++nt)
        brow[nt] = wt + (size_t)(nbase + nt * 16 + l16) * D_MODEL;

    v8f acc[2][4];
#pragma unroll
    for (int mt = 0; mt < 2; ++mt)
#pragma unroll
        for (int nt = 0; nt < 4; ++nt) acc[mt][nt] = vzero8();

    for (int k = 0; k < D_MODEL; k += 32) {
        v16bf a[2], b[4];
#pragma unroll
        for (int mt = 0; mt < 2; ++mt)
            a[mt] = ld_frag(arow[mt] + k + hh * 16, arow[mt] + k + hh * 16 + 8);
#pragma unroll
        for (int nt = 0; nt < 4; ++nt)
            b[nt] = ld_frag(brow[nt] + k + hh * 16, brow[nt] + k + hh * 16 + 8);
#pragma unroll
        for (int mt = 0; mt < 2; ++mt)
#pragma unroll
            for (int nt = 0; nt < 4; ++nt)
                acc[mt][nt] = WMMA_BF16(a[mt], b[nt], acc[mt][nt]);
    }

    if (nbase < 4096) {                       // uniform: whole wave writes Q|K
#pragma unroll
        for (int mt = 0; mt < 2; ++mt)
#pragma unroll
            for (int nt = 0; nt < 4; ++nt) {
                const int col = nbase + nt * 16 + l16;
                const float bv = bias[col];
#pragma unroll
                for (int r = 0; r < 8; ++r) {
                    const int row = mbase + mt * 16 + r + 8 * hh;
                    qk[(size_t)row * 4096 + col] = f2bf(acc[mt][nt][r] + bv);
                }
            }
    } else {                                   // uniform: whole wave writes V^T
#pragma unroll
        for (int mt = 0; mt < 2; ++mt)
#pragma unroll
            for (int nt = 0; nt < 4; ++nt) {
                const int feat = nbase - 4096 + nt * 16 + l16;
                const float bv = bias[nbase + nt * 16 + l16];
#pragma unroll
                for (int r = 0; r < 8; ++r) {
                    const int row = mbase + mt * 16 + r + 8 * hh;
                    vt[(size_t)feat * SEQ + row] = f2bf(acc[mt][nt][r] + bv);
                }
            }
    }
}

// ---------------------------------------------------------------------------
// Flash attention, causal, no 1/sqrt(d) scale (faithful to reference).
// grid = (seq/128, NHEADS); 8 waves/block; each wave owns a 16-row q tile.
// kv tiles of 64; per step: S = Q K^T (16 WMMA), online softmax, P V (16 WMMA).
// Tiles strictly below the diagonal skip the causal-mask compares.
// ---------------------------------------------------------------------------
__global__ __launch_bounds__(256) void attn_kernel(const unsigned short* __restrict__ qk,
                                                   const unsigned short* __restrict__ vt,
                                                   unsigned short* __restrict__ obf) {
    __shared__ __align__(16) unsigned short plds[8][16 * 64];

    const int lane = threadIdx.x & 31;
    const int wave = threadIdx.x >> 5;
    const int l16  = lane & 15;
    const int hh   = lane >> 4;
    const int head = blockIdx.y;
    const int qbase = blockIdx.x * 128 + wave * 16;

    // Q fragments (16 x 128), resident for the whole kernel.
    const unsigned short* qrow = qk + (size_t)(qbase + l16) * 4096 + head * DHEAD;
    v16bf qf[4];
#pragma unroll
    for (int f = 0; f < 4; ++f)
        qf[f] = ld_frag(qrow + f * 32 + hh * 16, qrow + f * 32 + hh * 16 + 8);

    v8f o[8];
#pragma unroll
    for (int d = 0; d < 8; ++d) o[d] = vzero8();
    float rmax[8], rsum[8];
#pragma unroll
    for (int r = 0; r < 8; ++r) { rmax[r] = -__builtin_inff(); rsum[r] = 0.0f; }

    unsigned short* my = plds[wave];
    const int kv_steps = (qbase + 16 + 63) >> 6;   // causal extent, tiles of 64

    for (int step = 0; step < kv_steps; ++step) {
        const int kvb = step * 64;
        const bool on_diag = (kvb + 64 > qbase);   // uniform per wave

        // ---- S = Q K^T : 16x64 tile, K-dim = 128 ----
        v8f s[4];
#pragma unroll
        for (int ct = 0; ct < 4; ++ct) s[ct] = vzero8();
#pragma unroll
        for (int ct = 0; ct < 4; ++ct) {
            const unsigned short* krow =
                qk + (size_t)(kvb + ct * 16 + l16) * 4096 + D_MODEL + head * DHEAD;
#pragma unroll
            for (int f = 0; f < 4; ++f) {
                v16bf b = ld_frag(krow + f * 32 + hh * 16, krow + f * 32 + hh * 16 + 8);
                s[ct] = WMMA_BF16(qf[f], b, s[ct]);
            }
        }

        // ---- causal mask + online softmax (rows live in 16-lane halves) ----
        float p[4][8];
#pragma unroll
        for (int r = 0; r < 8; ++r) {
            const int qi = qbase + r + 8 * hh;
            float mx = -__builtin_inff();
#pragma unroll
            for (int ct = 0; ct < 4; ++ct) {
                float v = s[ct][r];
                if (on_diag) {
                    const int ki = kvb + ct * 16 + l16;
                    v = (ki <= qi) ? v : -__builtin_inff();
                }
                p[ct][r] = v;
                mx = fmaxf(mx, v);
            }
#pragma unroll
            for (int m = 1; m < 16; m <<= 1) mx = fmaxf(mx, __shfl_xor(mx, m, 32));
            const float nmax  = fmaxf(rmax[r], mx);
            const float scale = __expf(rmax[r] - nmax);
            rmax[r] = nmax;
            float lsum = 0.0f;
#pragma unroll
            for (int ct = 0; ct < 4; ++ct) {
                const float e = __expf(p[ct][r] - nmax);
                p[ct][r] = e;
                lsum += e;
            }
#pragma unroll
            for (int m = 1; m < 16; m <<= 1) lsum += __shfl_xor(lsum, m, 32);
            rsum[r] = rsum[r] * scale + lsum;
#pragma unroll
            for (int d = 0; d < 8; ++d) o[d][r] *= scale;
            // stage P (bf16) into per-wave LDS: [16 rows][64 kv]
#pragma unroll
            for (int ct = 0; ct < 4; ++ct)
                my[(r + 8 * hh) * 64 + ct * 16 + l16] = f2bf(p[ct][r]);
        }

        // ---- O += P V : A = P (16x64) from LDS, B = V^T rows (contiguous) ----
#pragma unroll
        for (int f2 = 0; f2 < 2; ++f2) {
            v16bf a = ld_frag(my + l16 * 64 + f2 * 32 + hh * 16,
                              my + l16 * 64 + f2 * 32 + hh * 16 + 8);
#pragma unroll
            for (int d = 0; d < 8; ++d) {
                const unsigned short* vrow =
                    vt + (size_t)(head * DHEAD + d * 16 + l16) * SEQ + kvb;
                v16bf b = ld_frag(vrow + f2 * 32 + hh * 16, vrow + f2 * 32 + hh * 16 + 8);
                o[d] = WMMA_BF16(a, b, o[d]);
            }
        }
    }

    // ---- normalize and store O (bf16, [seq, d_model]) ----
#pragma unroll
    for (int r = 0; r < 8; ++r) {
        const float inv = 1.0f / rsum[r];
        const size_t row = qbase + r + 8 * hh;
#pragma unroll
        for (int d = 0; d < 8; ++d)
            obf[row * D_MODEL + head * DHEAD + d * 16 + l16] = f2bf(o[d][r] * inv);
    }
}

// ---------------------------------------------------------------------------
// Output projection: out[2048,2048] = O @ W_out + b_out (f32 result).
// Same tiling as QKV GEMM; B given as W_out^T [2048,2048] bf16.
// ---------------------------------------------------------------------------
__global__ __launch_bounds__(256) void proj_gemm_kernel(const unsigned short* __restrict__ ab,
                                                        const unsigned short* __restrict__ wt,
                                                        const float* __restrict__ bias,
                                                        float* __restrict__ out) {
    const int lane  = threadIdx.x & 31;
    const int wave  = threadIdx.x >> 5;
    const int l16   = lane & 15;
    const int hh    = lane >> 4;
    const int waveM = wave >> 1;
    const int waveN = wave & 1;
    const int mbase = blockIdx.y * 128 + waveM * 32;
    const int nbase = blockIdx.x * 128 + waveN * 64;

    const unsigned short* arow[2];
    const unsigned short* brow[4];
#pragma unroll
    for (int mt = 0; mt < 2; ++mt)
        arow[mt] = ab + (size_t)(mbase + mt * 16 + l16) * D_MODEL;
#pragma unroll
    for (int nt = 0; nt < 4; ++nt)
        brow[nt] = wt + (size_t)(nbase + nt * 16 + l16) * D_MODEL;

    v8f acc[2][4];
#pragma unroll
    for (int mt = 0; mt < 2; ++mt)
#pragma unroll
        for (int nt = 0; nt < 4; ++nt) acc[mt][nt] = vzero8();

    for (int k = 0; k < D_MODEL; k += 32) {
        v16bf a[2], b[4];
#pragma unroll
        for (int mt = 0; mt < 2; ++mt)
            a[mt] = ld_frag(arow[mt] + k + hh * 16, arow[mt] + k + hh * 16 + 8);
#pragma unroll
        for (int nt = 0; nt < 4; ++nt)
            b[nt] = ld_frag(brow[nt] + k + hh * 16, brow[nt] + k + hh * 16 + 8);
#pragma unroll
        for (int mt = 0; mt < 2; ++mt)
#pragma unroll
            for (int nt = 0; nt < 4; ++nt)
                acc[mt][nt] = WMMA_BF16(a[mt], b[nt], acc[mt][nt]);
    }

#pragma unroll
    for (int mt = 0; mt < 2; ++mt)
#pragma unroll
        for (int nt = 0; nt < 4; ++nt) {
            const int col = nbase + nt * 16 + l16;
            const float bv = bias[col];
#pragma unroll
            for (int r = 0; r < 8; ++r) {
                const int row = mbase + mt * 16 + r + 8 * hh;
                out[(size_t)row * D_MODEL + col] = acc[mt][nt][r] + bv;
            }
        }
}

// ---------------------------------------------------------------------------
// Host-side orchestration (graph-capture safe: only kernel launches on stream)
// ---------------------------------------------------------------------------
extern "C" void kernel_launch(void* const* d_in, const int* in_sizes, int n_in,
                              void* d_out, int out_size, void* d_ws, size_t ws_size,
                              hipStream_t stream) {
    const float* x     = (const float*)d_in[0];   // [2048, 2048]
    const float* W_qkv = (const float*)d_in[1];   // [2048, 6144]
    const float* b_qkv = (const float*)d_in[2];   // [6144]
    const float* W_out = (const float*)d_in[3];   // [2048, 2048]
    const float* b_out = (const float*)d_in[4];   // [2048]
    float* out = (float*)d_out;

    char* ws = (char*)d_ws;
    const size_t MB = (size_t)1 << 20;
    unsigned short* xb    = (unsigned short*)(ws);             //  8 MiB  x  bf16
    unsigned short* wqkvT = (unsigned short*)(ws +  8 * MB);   // 24 MiB  Wqkv^T bf16
    unsigned short* woutT = (unsigned short*)(ws + 32 * MB);   //  8 MiB  Wout^T bf16
    unsigned short* qk    = (unsigned short*)(ws + 40 * MB);   // 16 MiB  [seq, 4096] Q|K
    unsigned short* vt    = (unsigned short*)(ws + 56 * MB);   //  8 MiB  V^T [2048, seq]
    unsigned short* obf   = (unsigned short*)(ws + 64 * MB);   //  8 MiB  attn out bf16

    // 1) activations -> bf16
    {
        const int n = SEQ * D_MODEL;                           // 4M elems
        cvt_bf16_kernel<<<(n / 4 + 255) / 256, 256, 0, stream>>>(x, xb, n);
    }
    // 2) weight transposes -> bf16 (B^T layout for contiguous WMMA B fragments)
    transpose_bf16_kernel<<<dim3(6144 / 32, 2048 / 32), 256, 0, stream>>>(W_qkv, wqkvT, 2048, 6144);
    transpose_bf16_kernel<<<dim3(2048 / 32, 2048 / 32), 256, 0, stream>>>(W_out, woutT, 2048, 2048);
    // 3) QKV projection (writes Q|K row-major, V transposed)
    qkv_gemm_kernel<<<dim3(6144 / 128, 2048 / 128), 256, 0, stream>>>(xb, wqkvT, b_qkv, qk, vt);
    // 4) causal flash attention
    attn_kernel<<<dim3(SEQ / 128, NHEADS), 256, 0, stream>>>(qk, vt, obf);
    // 5) output projection (+bias, f32 out)
    proj_gemm_kernel<<<dim3(2048 / 128, 2048 / 128), 256, 0, stream>>>(obf, woutT, b_out, out);
}
